// MultiHeadSelfAttention_3453153706221
// MI455X (gfx1250) — compile-verified, matrix operand order checked by reading, not run on hardware
//
#include <hip/hip_runtime.h>
#include <hip/hip_bf16.h>

typedef __attribute__((ext_vector_type(16))) _Float16 v16h;
typedef __attribute__((ext_vector_type(8)))  _Float16 v8h;
typedef __attribute__((ext_vector_type(8)))  float    v8f;
typedef __attribute__((ext_vector_type(4)))  int      v4i;

#define T_SEQ 2048
#define C_DIM 1024
#define H_NUM 16
#define HS    64

__device__ __forceinline__ v8f wmma16(v16h a, v16h b, v8f c) {
    // (neg_a, A, neg_b, B, c_mod, C, reuse_a, reuse_b)
    return __builtin_amdgcn_wmma_f32_16x16x32_f16(false, a, false, b, (short)0, c, false, false);
}

__device__ __forceinline__ v16h load_cvt16(const float* __restrict__ p) {
    v16h r;
#pragma unroll
    for (int i = 0; i < 16; ++i) r[i] = (_Float16)p[i];
    return r;
}

__device__ __forceinline__ v8h load_cvt8(const float* __restrict__ p) {
    v8h r;
#pragma unroll
    for (int i = 0; i < 8; ++i) r[i] = (_Float16)p[i];
    return r;
}

// ---- gfx1250 async global->LDS copy (ASYNCcnt-tracked), with fallback -----
#if __has_builtin(__builtin_amdgcn_global_load_async_to_lds_b128)
#define HAVE_ASYNC_LDS 1
__device__ __forceinline__ void async_cp_b128(const void* g, void* l) {
    __builtin_amdgcn_global_load_async_to_lds_b128(
        (__attribute__((address_space(1))) v4i*)g,
        (__attribute__((address_space(3))) v4i*)l, 0, 0);
}
__device__ __forceinline__ void wait_async0() {
#if __has_builtin(__builtin_amdgcn_s_wait_asynccnt)
    __builtin_amdgcn_s_wait_asynccnt(0);
#else
    asm volatile("s_wait_asynccnt 0" ::: "memory");
#endif
}
#endif

// copy 16 f16 (32B) global -> LDS
__device__ __forceinline__ void stage16h(const _Float16* g, _Float16* l) {
#if defined(HAVE_ASYNC_LDS)
    async_cp_b128(g, l);
    async_cp_b128(g + 8, l + 8);
#else
    *(v16h*)l = *(const v16h*)g;
#endif
}

__device__ __forceinline__ void stage_fence() {
#if defined(HAVE_ASYNC_LDS)
    wait_async0();
#endif
}

// ---------------------------------------------------------------------------
// QKV projection: out_f16[b,h,t,hs] = x[b,t,:] @ W^T + bias   (f32 in, f16 out)
// block tile 128(M) x 64(N); 8 waves, each a 32x32 register tile (2x2 WMMA)
// grid = (C/64, 8192/128) = (16, 64), block = 256
// ---------------------------------------------------------------------------
__global__ void __launch_bounds__(256) qkv_gemm(const float* __restrict__ X,
                                                const float* __restrict__ W,
                                                const float* __restrict__ bias,
                                                _Float16* __restrict__ out) {
    __shared__ __align__(64) _Float16 xs[128][32];
    __shared__ __align__(64) _Float16 ws_[64][32];

    const int tid  = threadIdx.x;
    const int lane = tid & 31;
    const int wave = tid >> 5;
    const int mw   = wave & 3;            // 4 waves along M
    const int nw   = wave >> 2;           // 2 waves along N
    const int ln15 = lane & 15;
    const int half = lane >> 4;
    const int K    = C_DIM;

    const int rowBase = blockIdx.y * 128;
    const int colBase = blockIdx.x * 64;

    v8f acc[2][2] = {{v8f{}, v8f{}}, {v8f{}, v8f{}}};

    // staging source pointers (per thread)
    const float* xsrc = X + (size_t)(rowBase + (tid >> 1)) * K + (tid & 1) * 16;
    const float* wsrc = W + (size_t)(colBase + (tid >> 2)) * K + (tid & 3) * 8;

    for (int k0 = 0; k0 < K; k0 += 32) {
        __builtin_prefetch(xsrc + k0 + 64, 0, 0);   // global_prefetch_b8
        // stage X tile 128x32 (f32 -> f16), 16 elems/thread
        *(v16h*)&xs[tid >> 1][(tid & 1) * 16] = load_cvt16(xsrc + k0);
        // stage W tile 64x32 (f32 -> f16), 8 elems/thread
        *(v8h*)&ws_[tid >> 2][(tid & 3) * 8] = load_cvt8(wsrc + k0);
        __syncthreads();

        v16h a0 = *(const v16h*)&xs[mw * 32 + ln15][half * 16];
        v16h a1 = *(const v16h*)&xs[mw * 32 + 16 + ln15][half * 16];
        v16h b0 = *(const v16h*)&ws_[nw * 32 + ln15][half * 16];
        v16h b1 = *(const v16h*)&ws_[nw * 32 + 16 + ln15][half * 16];
        acc[0][0] = wmma16(a0, b0, acc[0][0]);
        acc[0][1] = wmma16(a0, b1, acc[0][1]);
        acc[1][0] = wmma16(a1, b0, acc[1][0]);
        acc[1][1] = wmma16(a1, b1, acc[1][1]);
        __syncthreads();
    }

    const int half8 = half * 8;
#pragma unroll
    for (int nt = 0; nt < 2; ++nt) {
        const int col = colBase + nw * 32 + nt * 16 + ln15;
        const float bv = bias[col];
        const int h = col >> 6, hs = col & 63;
#pragma unroll
        for (int mt = 0; mt < 2; ++mt) {
#pragma unroll
            for (int r = 0; r < 8; ++r) {
                int grow = rowBase + mw * 32 + mt * 16 + r + half8;  // 0..8191
                int b_ = grow >> 11;
                int t_ = grow & (T_SEQ - 1);
                size_t oidx = (((size_t)b_ * H_NUM + h) * T_SEQ + t_) * HS + hs;
                out[oidx] = (_Float16)(acc[mt][nt][r] + bv);
            }
        }
    }
}

// ---------------------------------------------------------------------------
// Causal flash attention: Q,K,V f16 [B,H,T,HS] -> attout f16 [B*T, C]
// grid = (T/64, B*H), block = 128 (4 waves, 16 q-rows each)
// ---------------------------------------------------------------------------
__global__ void __launch_bounds__(128) attn(const _Float16* __restrict__ Q,
                                            const _Float16* __restrict__ Kd,
                                            const _Float16* __restrict__ Vd,
                                            _Float16* __restrict__ attout) {
    __shared__ __align__(64) _Float16 ptile[4][16][32];
    __shared__ __align__(64) _Float16 vtile[32][HS + 8];   // 144B rows (16B mult)

    const int lane  = threadIdx.x & 31;
    const int wave  = threadIdx.x >> 5;
    const int bh    = blockIdx.y;
    const int qb0   = blockIdx.x * 64 + wave * 16;
    const int half  = lane >> 4;
    const int half8 = half * 8;
    const int ln15  = lane & 15;

    const _Float16* Qh = Q  + (size_t)bh * T_SEQ * HS;
    const _Float16* Kh = Kd + (size_t)bh * T_SEQ * HS;
    const _Float16* Vh = Vd + (size_t)bh * T_SEQ * HS;

    const v16h qa0 = *(const v16h*)(Qh + (size_t)(qb0 + ln15) * HS + half * 16);
    const v16h qa1 = *(const v16h*)(Qh + (size_t)(qb0 + ln15) * HS + 32 + half * 16);

    v8f oacc[4] = {v8f{}, v8f{}, v8f{}, v8f{}};
    float m_r[8], l_r[8];
#pragma unroll
    for (int r = 0; r < 8; ++r) { m_r[r] = -1e30f; l_r[r] = 0.0f; }

    const float scale = 0.125f;                 // 1/sqrt(64)
    const int kend = (blockIdx.x + 1) * 64;     // uniform across waves

    // V staging coordinates: 32 rows x 64 cols, 16 f16 per thread
    const int vrow = threadIdx.x >> 2;          // 0..31
    const int vcol = (threadIdx.x & 3) * 16;    // 0,16,32,48

    for (int kb = 0; kb < kend; kb += 32) {
        __syncthreads();                         // WAR: prior reads of vtile done
        stage16h(Vh + (size_t)(kb + vrow) * HS + vcol, &vtile[vrow][vcol]);
        stage_fence();                           // s_wait_asynccnt 0
        __syncthreads();

        // S = Q K^T : two 16x16 key tiles, contraction HS=64 = 2 x k32
        v16h kf0a = *(const v16h*)(Kh + (size_t)(kb + ln15) * HS + half * 16);
        v16h kf0b = *(const v16h*)(Kh + (size_t)(kb + ln15) * HS + 32 + half * 16);
        v16h kf1a = *(const v16h*)(Kh + (size_t)(kb + 16 + ln15) * HS + half * 16);
        v16h kf1b = *(const v16h*)(Kh + (size_t)(kb + 16 + ln15) * HS + 32 + half * 16);
        v8f s0 = {}, s1 = {};
        s0 = wmma16(qa0, kf0a, s0);
        s0 = wmma16(qa1, kf0b, s0);
        s1 = wmma16(qa0, kf1a, s1);
        s1 = wmma16(qa1, kf1b, s1);

        // online softmax over this 32-key slab
#pragma unroll
        for (int r = 0; r < 8; ++r) {
            const int qrow = qb0 + r + half8;
            const int key0 = kb + ln15;
            const int key1 = key0 + 16;
            float x0 = (key0 <= qrow) ? s0[r] * scale : -1e30f;
            float x1 = (key1 <= qrow) ? s1[r] * scale : -1e30f;
            float mx = fmaxf(x0, x1);
#pragma unroll
            for (int off = 1; off < 16; off <<= 1) mx = fmaxf(mx, __shfl_xor(mx, off));
            float mnew  = fmaxf(m_r[r], mx);
            float alpha = __expf(m_r[r] - mnew);
            float p0 = __expf(x0 - mnew);
            float p1 = __expf(x1 - mnew);
            float ps = p0 + p1;
#pragma unroll
            for (int off = 1; off < 16; off <<= 1) ps += __shfl_xor(ps, off);
            l_r[r] = l_r[r] * alpha + ps;
            m_r[r] = mnew;
#pragma unroll
            for (int dt = 0; dt < 4; ++dt) oacc[dt][r] *= alpha;
            ptile[wave][r + half8][ln15]      = (_Float16)p0;
            ptile[wave][r + half8][16 + ln15] = (_Float16)p1;
        }

        // O += P V
        v16h pa = *(const v16h*)&ptile[wave][ln15][half * 16];
#pragma unroll
        for (int dt = 0; dt < 4; ++dt) {
            v16h vf;
            const int d  = dt * 16 + ln15;
            const int ks = half * 16;
#pragma unroll
            for (int j = 0; j < 16; ++j) vf[j] = vtile[ks + j][d];
            oacc[dt] = wmma16(pa, vf, oacc[dt]);
        }
    }

    const int b_ = bh >> 4, h_ = bh & 15;
#pragma unroll
    for (int dt = 0; dt < 4; ++dt) {
        const int col = h_ * HS + dt * 16 + ln15;
#pragma unroll
        for (int r = 0; r < 8; ++r) {
            const int t_ = qb0 + r + half8;
            attout[((size_t)b_ * T_SEQ + t_) * C_DIM + col] =
                (_Float16)(oacc[dt][r] / l_r[r]);
        }
    }
}

// ---------------------------------------------------------------------------
// Output projection: y_f32[8192,1024] = attout_f16 @ Wp^T + bp
// same tiling as qkv_gemm; A tile staged via async global->LDS
// ---------------------------------------------------------------------------
__global__ void __launch_bounds__(256) proj_gemm(const _Float16* __restrict__ A,
                                                 const float* __restrict__ W,
                                                 const float* __restrict__ bias,
                                                 float* __restrict__ out) {
    __shared__ __align__(64) _Float16 as_[128][32];
    __shared__ __align__(64) _Float16 ws_[64][32];

    const int tid  = threadIdx.x;
    const int lane = tid & 31;
    const int wave = tid >> 5;
    const int mw   = wave & 3;
    const int nw   = wave >> 2;
    const int ln15 = lane & 15;
    const int half = lane >> 4;
    const int K    = C_DIM;

    const int rowBase = blockIdx.y * 128;
    const int colBase = blockIdx.x * 64;

    v8f acc[2][2] = {{v8f{}, v8f{}}, {v8f{}, v8f{}}};

    const _Float16* asrc = A + (size_t)(rowBase + (tid >> 1)) * K + (tid & 1) * 16;
    const float*    wsrc = W + (size_t)(colBase + (tid >> 2)) * K + (tid & 3) * 8;

    for (int k0 = 0; k0 < K; k0 += 32) {
        __builtin_prefetch(asrc + k0 + 64, 0, 0);
        stage16h(asrc + k0, &as_[tid >> 1][(tid & 1) * 16]);    // async -> LDS
        *(v8h*)&ws_[tid >> 2][(tid & 3) * 8] = load_cvt8(wsrc + k0);
        stage_fence();
        __syncthreads();

        v16h a0 = *(const v16h*)&as_[mw * 32 + ln15][half * 16];
        v16h a1 = *(const v16h*)&as_[mw * 32 + 16 + ln15][half * 16];
        v16h b0 = *(const v16h*)&ws_[nw * 32 + ln15][half * 16];
        v16h b1 = *(const v16h*)&ws_[nw * 32 + 16 + ln15][half * 16];
        acc[0][0] = wmma16(a0, b0, acc[0][0]);
        acc[0][1] = wmma16(a0, b1, acc[0][1]);
        acc[1][0] = wmma16(a1, b0, acc[1][0]);
        acc[1][1] = wmma16(a1, b1, acc[1][1]);
        __syncthreads();
    }

    const int half8 = half * 8;
#pragma unroll
    for (int nt = 0; nt < 2; ++nt) {
        const int col = colBase + nw * 32 + nt * 16 + ln15;
        const float bv = bias[col];
#pragma unroll
        for (int mt = 0; mt < 2; ++mt) {
#pragma unroll
            for (int r = 0; r < 8; ++r) {
                int grow = rowBase + mw * 32 + mt * 16 + r + half8;
                out[(size_t)grow * C_DIM + col] = acc[mt][nt][r] + bv;
            }
        }
    }
}

// ---------------------------------------------------------------------------
extern "C" void kernel_launch(void* const* d_in, const int* in_sizes, int n_in,
                              void* d_out, int out_size, void* d_ws, size_t ws_size,
                              hipStream_t stream) {
    const float* x  = (const float*)d_in[0];
    const float* Wq = (const float*)d_in[1];
    const float* bq = (const float*)d_in[2];
    const float* Wk = (const float*)d_in[3];
    const float* bk = (const float*)d_in[4];
    const float* Wv = (const float*)d_in[5];
    const float* bv = (const float*)d_in[6];
    const float* Wp = (const float*)d_in[7];
    const float* bp = (const float*)d_in[8];
    float* y = (float*)d_out;

    // workspace: 4 x 8M f16 elements (Q,K,V [B,H,T,HS], attout [B*T,C]) = 64MB
    const size_t NELT = (size_t)4 * H_NUM * T_SEQ * HS;
    _Float16* qbuf = (_Float16*)d_ws;
    _Float16* kbuf = qbuf + NELT;
    _Float16* vbuf = kbuf + NELT;
    _Float16* abuf = vbuf + NELT;

    dim3 gemmGrid(C_DIM / 64, (4 * T_SEQ) / 128);   // (16, 64)
    dim3 gemmBlk(256);
    qkv_gemm<<<gemmGrid, gemmBlk, 0, stream>>>(x, Wq, bq, qbuf);
    qkv_gemm<<<gemmGrid, gemmBlk, 0, stream>>>(x, Wk, bk, kbuf);
    qkv_gemm<<<gemmGrid, gemmBlk, 0, stream>>>(x, Wv, bv, vbuf);

    dim3 attnGrid(T_SEQ / 64, 4 * H_NUM);           // (32, 64)
    attn<<<attnGrid, dim3(128), 0, stream>>>(qbuf, kbuf, vbuf, abuf);

    proj_gemm<<<gemmGrid, gemmBlk, 0, stream>>>(abuf, Wp, bp, y);
}
